// Encoder_49684181680833
// MI455X (gfx1250) — matrix operand order repaired and checked by reading, not compiled, not run on hardware
//
#include <hip/hip_runtime.h>
#include <hip/hip_bf16.h>

// 3-layer LSTM encoder for MI455X (gfx1250), wave32 + bf16 WMMA.
// Per-layer persistent-tile kernel: combined [Wih;Whh] weights live in LDS
// (pre-swizzled to WMMA B-fragment layout), cell state c lives in VGPRs,
// h is exchanged through the LDS A-panel each timestep.
// Gate activations use gfx1250's hardware v_tanh_f32 when available.

typedef __attribute__((ext_vector_type(16))) __bf16        v16bf;
typedef __attribute__((ext_vector_type(8)))  float         v8f;
typedef __attribute__((ext_vector_type(4)))  unsigned int  v4u;
typedef __attribute__((ext_vector_type(8)))  unsigned int  v8u;

#define DEV_INLINE __device__ __forceinline__

static constexpr int T_SEQ    = 512;
static constexpr int BATCH    = 512;
static constexpr int BTILE    = 64;
static constexpr int NTHREADS = 256;   // 8 wave32s

DEV_INLINE unsigned short f32_to_bf16_rne(float f) {
  unsigned int u = __builtin_bit_cast(unsigned int, f);
  u += 0x7FFFu + ((u >> 16) & 1u);
  return (unsigned short)(u >> 16);
}

#if __has_builtin(__builtin_amdgcn_tanhf)
DEV_INLINE float fast_tanh(float x) { return __builtin_amdgcn_tanhf(x); }
#else
DEV_INLINE float fast_tanh(float x) {
  // tanh(x) = 2/(1+exp(-2x)) - 1
  return 2.0f / (1.0f + __expf(-2.0f * x)) - 1.0f;
}
#endif
DEV_INLINE float fast_sigmoid(float x) { return fmaf(0.5f, fast_tanh(0.5f * x), 0.5f); }

// A-fragment (16x32 bf16): lane 0-15 -> row M=lane, K = k0+0..7 & k0+16..23;
// lane 16-31 -> row M=lane-16, K = k0+8..15 & k0+24..31.
DEV_INLINE v16bf load_frag_a(const unsigned short* p) {
  v4u lo = *(const v4u*)(p);
  v4u hi = *(const v4u*)(p + 16);
  v8u t;
  t[0] = lo[0]; t[1] = lo[1]; t[2] = lo[2]; t[3] = lo[3];
  t[4] = hi[0]; t[5] = hi[1]; t[6] = hi[2]; t[7] = hi[3];
  return __builtin_bit_cast(v16bf, t);
}

// B-fragment: lane half h reads K = 8h+0..7 & 16+8h+0..7 of the 32-wide k-row.
DEV_INLINE v16bf load_frag_b(const unsigned short* p) {
  return load_frag_a(p);
}

template <int K_IN, int K_IN_PAD, int H, bool FINAL, bool XF32>
__global__ __launch_bounds__(NTHREADS, 1)
void lstm_layer_kernel(const void* __restrict__ xseq,
                       const float* __restrict__ Wih,
                       const float* __restrict__ Whh,
                       const float* __restrict__ bih,
                       const float* __restrict__ bhh,
                       unsigned short* __restrict__ yout,  // bf16 [B,T,H]
                       float* __restrict__ dout)           // f32  [B,H] (FINAL)
{
  constexpr int KTOT  = K_IN_PAD + H;       // combined K (multiple of 32)
  constexpr int FOURH = 4 * H;
  constexpr int KT    = KTOT / 32;          // k-tiles
  constexpr int HC    = H / 16;             // h column tiles
  constexpr int PPW   = (4 * HC) / (NTHREADS / 32);  // (m,hcol) pairs per wave

  extern __shared__ char smem_raw[];
  unsigned short* Wb  = (unsigned short*)smem_raw;            // [KT][4H][32] bf16
  unsigned short* As  = Wb + (size_t)KTOT * FOURH;            // [BTILE][KTOT] bf16
  float* biasLds      = (float*)(As + (size_t)BTILE * KTOT);  // [4H]

  const int tid   = threadIdx.x;
  const int wave  = tid >> 5;
  const int lane  = tid & 31;
  const int lhalf = lane >> 4;
  const int lcol  = lane & 15;
  const int b0    = blockIdx.x * BTILE;

  // ---- One-time staging: combined weights, transposed + padded, into LDS ----
  for (int i = tid; i < KTOT * FOURH; i += NTHREADS) {
    const int kl   = i & 31;
    const int rest = i >> 5;
    const int n    = rest % FOURH;
    const int kt   = rest / FOURH;
    const int k    = kt * 32 + kl;
    float w;
    if (k < K_IN)          w = Wih[(size_t)n * K_IN + k];
    else if (k < K_IN_PAD) w = 0.0f;
    else                   w = Whh[(size_t)n * H + (k - K_IN_PAD)];
    Wb[i] = f32_to_bf16_rne(w);
  }
  for (int n = tid; n < FOURH; n += NTHREADS) biasLds[n] = bih[n] + bhh[n];
  // zero pad columns [K_IN, K_IN_PAD) and h region [K_IN_PAD, KTOT) (h0 = 0)
  for (int i = tid; i < BTILE * (KTOT - K_IN); i += NTHREADS) {
    const int row = i / (KTOT - K_IN);
    const int k   = K_IN + i % (KTOT - K_IN);
    As[row * KTOT + k] = 0;
  }

  const v8f vzero = {0.f, 0.f, 0.f, 0.f, 0.f, 0.f, 0.f, 0.f};
  v8f c[PPW], hreg[PPW];
#pragma unroll
  for (int p = 0; p < PPW; ++p) { c[p] = vzero; hreg[p] = vzero; }

#pragma unroll 1
  for (int t = 0; t < T_SEQ; ++t) {
    // ---- stage x_t into A-panel columns [0, K_IN) ----
    if constexpr (XF32) {
      const float* xs = (const float*)xseq;
      for (int i = tid; i < BTILE * K_IN; i += NTHREADS) {
        const int row = i / K_IN, k = i % K_IN;
        As[row * KTOT + k] =
            f32_to_bf16_rne(xs[((size_t)(b0 + row) * T_SEQ + t) * K_IN + k]);
      }
      if (t + 1 < T_SEQ)
        __builtin_prefetch(
            xs + ((size_t)(b0 + (tid & (BTILE - 1))) * T_SEQ + (t + 1)) * K_IN, 0, 3);
    } else {
      // bf16 activations: copy 8 elements (16 B) at a time; both sides aligned.
      const unsigned short* xs = (const unsigned short*)xseq;
      constexpr int CHUNKS = BTILE * K_IN / 8;       // 16B chunks
      constexpr int CPR    = K_IN / 8;               // chunks per row
      for (int i = tid; i < CHUNKS; i += NTHREADS) {
        const int row = i / CPR, cc = i % CPR;
        const v4u v = *(const v4u*)(xs + ((size_t)(b0 + row) * T_SEQ + t) * K_IN + cc * 8);
        *(v4u*)(As + row * KTOT + cc * 8) = v;
      }
      if (t + 1 < T_SEQ)
        __builtin_prefetch(
            xs + ((size_t)(b0 + (tid & (BTILE - 1))) * T_SEQ + (t + 1)) * K_IN, 0, 3);
    }
    __syncthreads();  // x_t and h_{t-1} visible in A-panel

    // ---- gates = [x_t | h] @ Wc  (bf16 WMMA, f32 accumulate) ----
#pragma unroll
    for (int p = 0; p < PPW; ++p) {
      const int pairId = wave * PPW + p;
      const int mtile  = pairId / HC;
      const int hcol   = pairId % HC;

      const unsigned short* arow =
          As + (size_t)(mtile * 16 + lcol) * KTOT + lhalf * 8;
      auto bptr = [&](int kt, int g) {
        return Wb + ((size_t)kt * FOURH + g * H + hcol * 16 + lcol) * 32 + lhalf * 8;
      };

      v8f acc0 = vzero, acc1 = vzero, acc2 = vzero, acc3 = vzero;

      // software pipeline: fragments for kt+1 load under kt's WMMA quad
      v16bf a_c  = load_frag_a(arow);
      v16bf bI_c = load_frag_b(bptr(0, 0));
      v16bf bF_c = load_frag_b(bptr(0, 1));
      v16bf bG_c = load_frag_b(bptr(0, 2));
      v16bf bO_c = load_frag_b(bptr(0, 3));
#pragma unroll
      for (int kt = 0; kt < KT; ++kt) {
        v16bf a_n = a_c, bI_n = bI_c, bF_n = bF_c, bG_n = bG_c, bO_n = bO_c;
        if (kt + 1 < KT) {
          a_n  = load_frag_a(arow + (kt + 1) * 32);
          bI_n = load_frag_b(bptr(kt + 1, 0));
          bF_n = load_frag_b(bptr(kt + 1, 1));
          bG_n = load_frag_b(bptr(kt + 1, 2));
          bO_n = load_frag_b(bptr(kt + 1, 3));
        }
        acc0 = __builtin_amdgcn_wmma_f32_16x16x32_bf16(false, a_c, false, bI_c, (short)0, acc0, false, false);
        acc1 = __builtin_amdgcn_wmma_f32_16x16x32_bf16(false, a_c, false, bF_c, (short)0, acc1, false, false);
        acc2 = __builtin_amdgcn_wmma_f32_16x16x32_bf16(false, a_c, false, bG_c, (short)0, acc2, false, false);
        acc3 = __builtin_amdgcn_wmma_f32_16x16x32_bf16(false, a_c, false, bO_c, (short)0, acc3, false, false);
        a_c = a_n; bI_c = bI_n; bF_c = bF_n; bG_c = bG_n; bO_c = bO_n;
      }

      // ---- elementwise LSTM cell update, c kept in VGPRs ----
      const int   ncol = hcol * 16 + lcol;
      const float bI = biasLds[0 * H + ncol];
      const float bF = biasLds[1 * H + ncol];
      const float bG = biasLds[2 * H + ncol];
      const float bO = biasLds[3 * H + ncol];
#pragma unroll
      for (int r = 0; r < 8; ++r) {
        const float ig = fast_sigmoid(acc0[r] + bI);
        const float fg = fast_sigmoid(acc1[r] + bF);
        const float gg = fast_tanh(acc2[r] + bG);
        const float og = fast_sigmoid(acc3[r] + bO);
        const float cn = fmaf(fg, c[p][r], ig * gg);
        c[p][r]    = cn;
        hreg[p][r] = og * fast_tanh(cn);
      }
    }
    __syncthreads();  // all reads of old h complete before overwrite

    // ---- write h_t back to A-panel (bf16) and stream to global ----
#pragma unroll
    for (int p = 0; p < PPW; ++p) {
      const int pairId = wave * PPW + p;
      const int mtile  = pairId / HC;
      const int hcol   = pairId % HC;
#pragma unroll
      for (int r = 0; r < 8; ++r) {
        const int row = mtile * 16 + lhalf * 8 + r;  // C-tile layout row
        const int col = hcol * 16 + lcol;
        const float hv = hreg[p][r];
        const unsigned short hb = f32_to_bf16_rne(hv);
        As[row * KTOT + K_IN_PAD + col] = hb;
        if constexpr (!FINAL) {
          yout[((size_t)(b0 + row) * T_SEQ + t) * H + col] = hb;
        } else {
          if (t == T_SEQ - 1)
            dout[(size_t)(b0 + row) * H + col] = hv;
        }
      }
    }
  }
}

extern "C" void kernel_launch(void* const* d_in, const int* in_sizes, int n_in,
                              void* d_out, int out_size, void* d_ws, size_t ws_size,
                              hipStream_t stream) {
  (void)in_sizes; (void)n_in; (void)out_size; (void)ws_size;

  const float* x    = (const float*)d_in[0];
  const float* Wih1 = (const float*)d_in[1];
  const float* Whh1 = (const float*)d_in[2];
  const float* bih1 = (const float*)d_in[3];
  const float* bhh1 = (const float*)d_in[4];
  const float* Wih2 = (const float*)d_in[5];
  const float* Whh2 = (const float*)d_in[6];
  const float* bih2 = (const float*)d_in[7];
  const float* bhh2 = (const float*)d_in[8];
  const float* Wih3 = (const float*)d_in[9];
  const float* Whh3 = (const float*)d_in[10];
  const float* bih3 = (const float*)d_in[11];
  const float* bhh3 = (const float*)d_in[12];

  // Workspace: bf16 inter-layer activations (2 x 64 MiB).
  unsigned short* y1 = (unsigned short*)d_ws;
  unsigned short* y2 = y1 + (size_t)BATCH * T_SEQ * 128;

  // Dynamic LDS: Wb + A-panel (bf16) + bias (f32).
  constexpr int SM1 = (160 * 512 + BTILE * 160) * 2 + 512 * 4;  // 186368 B
  constexpr int SM2 = (256 * 512 + BTILE * 256) * 2 + 512 * 4;  // 296960 B
  constexpr int SM3 = (192 * 256 + BTILE * 192) * 2 + 256 * 4;  // 123904 B

  (void)hipFuncSetAttribute(
      reinterpret_cast<const void*>(&lstm_layer_kernel<8, 32, 128, false, true>),
      hipFuncAttributeMaxDynamicSharedMemorySize, SM1);
  (void)hipFuncSetAttribute(
      reinterpret_cast<const void*>(&lstm_layer_kernel<128, 128, 128, false, false>),
      hipFuncAttributeMaxDynamicSharedMemorySize, SM2);
  (void)hipFuncSetAttribute(
      reinterpret_cast<const void*>(&lstm_layer_kernel<128, 128, 64, true, false>),
      hipFuncAttributeMaxDynamicSharedMemorySize, SM3);

  const dim3 grid(BATCH / BTILE), block(NTHREADS);

  lstm_layer_kernel<8, 32, 128, false, true><<<grid, block, SM1, stream>>>(
      x, Wih1, Whh1, bih1, bhh1, y1, nullptr);
  lstm_layer_kernel<128, 128, 128, false, false><<<grid, block, SM2, stream>>>(
      y1, Wih2, Whh2, bih2, bhh2, y2, nullptr);
  lstm_layer_kernel<128, 128, 64, true, false><<<grid, block, SM3, stream>>>(
      y2, Wih3, Whh3, bih3, bhh3, nullptr, (float*)d_out);
}
